// BahdanauAttention_14654428414661
// MI455X (gfx1250) — matrix operand order repaired and checked by reading, not compiled, not run on hardware
//
#include <hip/hip_runtime.h>
#include <hip/hip_bf16.h>

#define FEAT   2048
#define HID    1024
#define ATTN   512
#define BATCH  128
#define NPOS   196
#define MTOT   (BATCH * NPOS)   // 25088 = 784 * 32

typedef __attribute__((ext_vector_type(16))) __bf16 v16bf;
typedef __attribute__((ext_vector_type(8)))  float  v8f;

union AFrag { v16bf v; uint4 q[2]; };
union BFrag { v16bf v; uint4 q[2]; };

// round-to-nearest-even f32 -> bf16, packed pair
__device__ __forceinline__ unsigned pack_bf16x2(float a, float b) {
    union { float f; unsigned u; } ua, ub;
    ua.f = a; ub.f = b;
    unsigned x = ua.u + (0x7FFFu + ((ua.u >> 16) & 1u));
    unsigned y = ub.u + (0x7FFFu + ((ub.u >> 16) & 1u));
    return (x >> 16) | (y & 0xFFFF0000u);
}

__device__ __forceinline__ BFrag load_bfrag(const unsigned* bp) {
    BFrag b;
    b.q[0] = *(const uint4*)bp;
    b.q[1] = *(const uint4*)(bp + 8);
    return b;
}

// CDNA5 async copy: 16 bytes global -> LDS, tracked by ASYNCcnt (no VGPR data path)
__device__ __forceinline__ void async_copy_b128(unsigned lds_off, const unsigned* gaddr) {
    asm volatile("global_load_async_to_lds_b128 %0, %1, off"
                 :: "v"(lds_off), "v"(gaddr)
                 : "memory");
}
__device__ __forceinline__ void wait_asynccnt0() {
    asm volatile("s_wait_asynccnt 0x0" ::: "memory");
}

// ---------------------------------------------------------------- K0a: We -> bf16
__global__ __launch_bounds__(256)
void convert_we_kernel(const float* __restrict__ We, unsigned* __restrict__ WeBf) {
    const int npairs = ATTN * FEAT / 2;
    int i = blockIdx.x * blockDim.x + threadIdx.x;
    if (i < npairs) {
        float2 f = ((const float2*)We)[i];
        WeBf[i] = pack_bf16x2(f.x, f.y);
    }
}

// ---------------------------------------------------------------- K0b: encoder_out -> bf16 (async path only)
__global__ __launch_bounds__(256)
void convert_enc_kernel(const float* __restrict__ enc, unsigned* __restrict__ encBf) {
    const long long npairs = (long long)MTOT * (FEAT / 2);
    long long i = (long long)blockIdx.x * blockDim.x + threadIdx.x;
    if (i < npairs) {
        float2 f = ((const float2*)enc)[i];
        encBf[i] = pack_bf16x2(f.x, f.y);
    }
}

// ---------------------------------------------------------------- K1: dec = hidden @ Wd^T + bd
__global__ __launch_bounds__(256)
void dec_kernel(const float* __restrict__ hidden, const float* __restrict__ Wd,
                const float* __restrict__ bd, float* __restrict__ dec) {
    const int b = blockIdx.x;
    const int tid = threadIdx.x;
    __shared__ float sh[HID];
    for (int i = tid; i < HID; i += 256) sh[i] = hidden[b * HID + i];
    __syncthreads();
    for (int a = tid; a < ATTN; a += 256) {
        const float4* w = (const float4*)(Wd + (size_t)a * HID);
        float acc = 0.0f;
        #pragma unroll 4
        for (int j = 0; j < HID / 4; ++j) {
            float4 v = w[j];
            float4 h = *(const float4*)&sh[j * 4];
            acc += v.x * h.x + v.y * h.y + v.z * h.z + v.w * h.w;
        }
        dec[b * ATTN + a] = acc + bd[a];
    }
}

// ================================================================ K2 shared pieces
__device__ __forceinline__ void wmma_step(v8f (&c)[2][4], const AFrag& a0, const AFrag& a1,
                                          const unsigned* bbase) {
    BFrag bcur = load_bfrag(bbase);
    #pragma unroll
    for (int st = 0; st < 4; ++st) {
        BFrag bnxt;
        if (st < 3) bnxt = load_bfrag(bbase + (size_t)((st + 1) << 4) * (FEAT / 2));
        c[0][st] = __builtin_amdgcn_wmma_f32_16x16x32_bf16(
                       false, a0.v, false, bcur.v, (short)0, c[0][st], false, false);
        c[1][st] = __builtin_amdgcn_wmma_f32_16x16x32_bf16(
                       false, a1.v, false, bcur.v, (short)0, c[1][st], false, false);
        if (st < 3) bcur = bnxt;
    }
}

__device__ __forceinline__ void scores_epilogue(
        v8f (&c)[2][4], int mbase, int nb, int lane,
        const float* __restrict__ be, const float* __restrict__ dec,
        const float* __restrict__ Wf, const float* __restrict__ bfp,
        float* __restrict__ scores, float* sRed, int tid)
{
    float sums[2][8];
    #pragma unroll
    for (int ms = 0; ms < 2; ++ms)
        #pragma unroll
        for (int r = 0; r < 8; ++r) sums[ms][r] = 0.0f;

    const int wl8 = (lane >> 4) << 3;          // C/D layout: lanes 16-31 hold rows M+8
    #pragma unroll
    for (int st = 0; st < 4; ++st) {
        const int col = nb + (st << 4) + (lane & 15);
        const float bec = be[col];
        const float wfc = Wf[col];
        #pragma unroll
        for (int ms = 0; ms < 2; ++ms) {
            #pragma unroll
            for (int r = 0; r < 8; ++r) {
                const int m = mbase + ms * 16 + r + wl8;
                const int b = m / NPOS;
                const float v = c[ms][st][r] + bec + dec[b * ATTN + col];
                sums[ms][r] += tanhf(v) * wfc;
            }
        }
    }
    #pragma unroll
    for (int ms = 0; ms < 2; ++ms)
        #pragma unroll
        for (int r = 0; r < 8; ++r)
            atomicAdd(&sRed[ms * 16 + r + wl8], sums[ms][r]);   // ds_add_f32
    __syncthreads();
    if (tid < 32) scores[mbase + tid] = sRed[tid] + bfp[0];
}

// ---------------------------------------------------------------- K2a: async-staged variant
// A pre-packed bf16; tiles staged via global_load_async_to_lds_b128, pipelined one tile ahead:
// iteration k issues the copy for tile k+1, computes tile k, then waits + barrier.
__global__ __launch_bounds__(256)
void scores_kernel_async(const unsigned* __restrict__ encBf,  // [MTOT][FEAT/2] packed bf16
                         const unsigned* __restrict__ WeBf,   // [ATTN][FEAT/2] packed bf16
                         const float* __restrict__ be,
                         const float* __restrict__ dec,
                         const float* __restrict__ Wf,
                         const float* __restrict__ bfp,
                         float* __restrict__ scores)
{
    __shared__ unsigned short sA[2][32 * 32];  // A tile bf16 [32][32], double buffered (2 KB each)
    __shared__ float sRed[32];

    const int tid  = threadIdx.x;
    const int lane = tid & 31;
    const int wave = tid >> 5;
    const int mbase = blockIdx.x << 5;
    const int nb = wave << 6;

    if (tid < 32) sRed[tid] = 0.0f;

    v8f c[2][4] = {};

    const int arow  = lane & 15;
    const int ahalf = (lane >> 4) << 3;        // 0 or 8 (halfs)
    const int bkoff = (lane >> 4) << 2;        // 0 or 4 (uints)

    // staging: threads 0..127 each async-copy 16 bytes of the 32x32 bf16 tile
    const int srow = tid >> 2;                 // 0..31 (for tid<128)
    const int sseg = tid & 3;                  // 0..3, 16-byte segments
    const unsigned ldsRow = (unsigned)(uintptr_t)&sA[0][0] + srow * 64 + sseg * 16;
    const unsigned* gRow = encBf + (size_t)(mbase + srow) * (FEAT / 2) + sseg * 4;

    // prologue: stage tile 0
    if (tid < 128) {
        async_copy_b128(ldsRow, gRow);
        wait_asynccnt0();
    }
    __syncthreads();

    for (int kk = 0; kk < FEAT; kk += 32) {
        const int buf = (kk >> 5) & 1;

        // kick off next tile's copy into the other buffer (latency hidden behind WMMAs)
        if (kk + 32 < FEAT && tid < 128)
            async_copy_b128(ldsRow + (buf ^ 1) * 2048, gRow + ((kk + 32) >> 1));

        AFrag a0, a1;
        const unsigned short* ab0 = &sA[buf][arow * 32 + ahalf];
        const unsigned short* ab1 = &sA[buf][(arow + 16) * 32 + ahalf];
        a0.q[0] = *(const uint4*)ab0;
        a0.q[1] = *(const uint4*)(ab0 + 16);
        a1.q[0] = *(const uint4*)ab1;
        a1.q[1] = *(const uint4*)(ab1 + 16);

        const unsigned* bbase = WeBf + (size_t)(nb + (lane & 15)) * (FEAT / 2) + (kk >> 1) + bkoff;
        wmma_step(c, a0, a1, bbase);

        if (tid < 128) wait_asynccnt0();       // next tile landed
        __syncthreads();
    }
    scores_epilogue(c, mbase, nb, lane, be, dec, Wf, bfp, scores, sRed, tid);
}

// ---------------------------------------------------------------- K2b: fallback (convert in LDS staging)
// f32 A tile for step k+1 is loaded into registers before the barrier of step k.
__global__ __launch_bounds__(256)
void scores_kernel(const float* __restrict__ enc,
                   const unsigned* __restrict__ WeBf,
                   const float* __restrict__ be,
                   const float* __restrict__ dec,
                   const float* __restrict__ Wf,
                   const float* __restrict__ bfp,
                   float* __restrict__ scores)
{
    __shared__ unsigned short sA[2][32 * 32];
    __shared__ float sRed[32];

    const int tid  = threadIdx.x;
    const int lane = tid & 31;
    const int wave = tid >> 5;
    const int mbase = blockIdx.x << 5;
    const int nb = wave << 6;

    if (tid < 32) sRed[tid] = 0.0f;

    v8f c[2][4] = {};

    const int arow  = lane & 15;
    const int ahalf = (lane >> 4) << 3;
    const int bkoff = (lane >> 4) << 2;

    const int srow = tid >> 3;
    const int scol = (tid & 7) << 2;
    const float* srcRow = enc + (size_t)(mbase + srow) * FEAT + scol;

    float4 f = *(const float4*)srcRow;         // prologue: tile 0

    for (int kk = 0; kk < FEAT; kk += 32) {
        const int buf = (kk >> 5) & 1;

        uint2 p;
        p.x = pack_bf16x2(f.x, f.y);
        p.y = pack_bf16x2(f.z, f.w);
        *(uint2*)&sA[buf][srow * 32 + scol] = p;
        if (kk + 32 < FEAT)
            f = *(const float4*)(srcRow + kk + 32);   // next tile, overlapped with compute
        if (kk + 64 < FEAT)
            __builtin_prefetch(srcRow + kk + 64, 0, 1);
        __syncthreads();

        AFrag a0, a1;
        const unsigned short* ab0 = &sA[buf][arow * 32 + ahalf];
        const unsigned short* ab1 = &sA[buf][(arow + 16) * 32 + ahalf];
        a0.q[0] = *(const uint4*)ab0;
        a0.q[1] = *(const uint4*)(ab0 + 16);
        a1.q[0] = *(const uint4*)ab1;
        a1.q[1] = *(const uint4*)(ab1 + 16);

        const unsigned* bbase = WeBf + (size_t)(nb + (lane & 15)) * (FEAT / 2) + (kk >> 1) + bkoff;
        wmma_step(c, a0, a1, bbase);
    }
    scores_epilogue(c, mbase, nb, lane, be, dec, Wf, bfp, scores, sRed, tid);
}

// ---------------------------------------------------------------- K3: softmax over P + context
__global__ __launch_bounds__(256)
void softmax_context_kernel(const float* __restrict__ enc,
                            const float* __restrict__ scores,
                            float* __restrict__ ctx,
                            float* __restrict__ alpha)
{
    const int b = blockIdx.x;
    const int tid = threadIdx.x;
    __shared__ float sa[NPOS];
    __shared__ float red[2];

    if (tid < NPOS) sa[tid] = scores[b * NPOS + tid];
    __syncthreads();
    if (tid == 0) {
        float m = -1e30f;
        for (int p = 0; p < NPOS; ++p) m = fmaxf(m, sa[p]);
        red[0] = m;
    }
    __syncthreads();
    const float mx = red[0];
    if (tid < NPOS) sa[tid] = __expf(sa[tid] - mx);
    __syncthreads();
    if (tid == 0) {
        float s = 0.0f;
        for (int p = 0; p < NPOS; ++p) s += sa[p];
        red[1] = 1.0f / s;
    }
    __syncthreads();
    const float inv = red[1];
    if (tid < NPOS) {
        const float al = sa[tid] * inv;
        sa[tid] = al;
        alpha[b * NPOS + tid] = al;
    }
    __syncthreads();

    const int f0 = tid * 8;
    float a0 = 0, a1 = 0, a2 = 0, a3 = 0, a4 = 0, a5 = 0, a6 = 0, a7 = 0;
    const float* base = enc + (size_t)b * NPOS * FEAT + f0;
    for (int p = 0; p < NPOS; ++p) {
        const float al = sa[p];
        const float4 x0 = *(const float4*)(base + (size_t)p * FEAT);
        const float4 x1 = *(const float4*)(base + (size_t)p * FEAT + 4);
        a0 += al * x0.x; a1 += al * x0.y; a2 += al * x0.z; a3 += al * x0.w;
        a4 += al * x1.x; a5 += al * x1.y; a6 += al * x1.z; a7 += al * x1.w;
    }
    float* o = ctx + (size_t)b * FEAT + f0;
    float4 r0 = { a0, a1, a2, a3 };
    float4 r1 = { a4, a5, a6, a7 };
    *(float4*)o = r0;
    *(float4*)(o + 4) = r1;
}

// ---------------------------------------------------------------- launcher
extern "C" void kernel_launch(void* const* d_in, const int* in_sizes, int n_in,
                              void* d_out, int out_size, void* d_ws, size_t ws_size,
                              hipStream_t stream) {
    const float* encoder_out = (const float*)d_in[0];
    const float* hidden      = (const float*)d_in[1];
    const float* We          = (const float*)d_in[2];
    const float* be          = (const float*)d_in[3];
    const float* Wd          = (const float*)d_in[4];
    const float* bd          = (const float*)d_in[5];
    const float* Wf          = (const float*)d_in[6];
    const float* bfp         = (const float*)d_in[7];

    float* out   = (float*)d_out;
    float* ctx   = out;                       // [128][2048]
    float* alpha = out + BATCH * FEAT;        // [128][196]

    float*    dec    = (float*)d_ws;                          // 65536 f32
    float*    scores = dec + BATCH * ATTN;                    // 25088 f32
    unsigned* WeBf   = (unsigned*)(scores + MTOT);            // 524288 u32
    unsigned* encBf  = WeBf + ATTN * (FEAT / 2);              // 25690112 u32 (optional)

    const size_t base_ws = (size_t)(BATCH * ATTN + MTOT + ATTN * (FEAT / 2)) * 4;
    const size_t full_ws = base_ws + (size_t)MTOT * (FEAT / 2) * 4;

    const int npairs_we = ATTN * FEAT / 2;
    convert_we_kernel<<<(npairs_we + 255) / 256, 256, 0, stream>>>(We, WeBf);
    dec_kernel<<<BATCH, 256, 0, stream>>>(hidden, Wd, bd, dec);

    if (ws_size >= full_ws) {
        const long long npairs_enc = (long long)MTOT * (FEAT / 2);
        const int nblk = (int)((npairs_enc + 255) / 256);
        convert_enc_kernel<<<nblk, 256, 0, stream>>>(encoder_out, encBf);
        scores_kernel_async<<<MTOT / 32, 256, 0, stream>>>(encBf, WeBf, be, dec, Wf, bfp, scores);
    } else {
        scores_kernel<<<MTOT / 32, 256, 0, stream>>>(encoder_out, WeBf, be, dec, Wf, bfp, scores);
    }

    softmax_context_kernel<<<BATCH, 256, 0, stream>>>(encoder_out, scores, ctx, alpha);
}
